// GIN_31825707664041
// MI455X (gfx1250) — matrix operand order repaired and checked by reading, not compiled, not run on hardware
//
#include <hip/hip_runtime.h>
#include <hip/hip_bf16.h>

#define DMID   128
#define NB     64
#define BN_EPS 1e-5f
#define LDS_STRIDE 136   // halves per row: 272B, multiple of 16B
#define ROWS   32        // rows per block iteration (2 x 16-row WMMA tiles)

typedef __attribute__((ext_vector_type(16))) _Float16 v16h;
typedef __attribute__((ext_vector_type(8)))  _Float16 v8h;
typedef __attribute__((ext_vector_type(8)))  float    v8f;

// ---------------------------------------------------------------------------
// Edge scatter: agg[dst] += x[src].  One wave (32 lanes x float4) per edge.
// ---------------------------------------------------------------------------
__global__ void gin_scatter_add(const float* __restrict__ xin,
                                const int* __restrict__ ei,
                                float* __restrict__ agg, int numE) {
  int e = blockIdx.x * (blockDim.x >> 5) + (threadIdx.x >> 5);
  if (e >= numE) return;
  int lane = threadIdx.x & 31;
  int s = ei[e];            // edge_index row 0 (src)
  int d = ei[numE + e];     // edge_index row 1 (dst)
  const float4 v = *(const float4*)(xin + (size_t)s * DMID + lane * 4);
  float* ap = agg + (size_t)d * DMID + lane * 4;
  atomicAdd(ap + 0, v.x);
  atomicAdd(ap + 1, v.y);
  atomicAdd(ap + 2, v.z);
  atomicAdd(ap + 3, v.w);
}

// ---------------------------------------------------------------------------
// Fused GIN MLP: out = relu( relu(BN((x+agg)@W1 + b1)) @ W2 + b2 )
// 256 threads = 8 waves; block handles 32 rows (two 16-row WMMA tiles) per
// iteration.  Wave w owns output columns [16w,16w+16); W1/W2 fragments stay
// VGPR-resident across the grid-stride tile loop; two independent
// accumulator chains per wave interleave WMMA issue with LDS loads.
// ---------------------------------------------------------------------------
__global__ void __launch_bounds__(256)
gin_mlp_wmma(const float* __restrict__ in,  const float* __restrict__ agg,
             const float* __restrict__ w1,  const float* __restrict__ b1,
             const float* __restrict__ gamma, const float* __restrict__ beta,
             const float* __restrict__ mean,  const float* __restrict__ var,
             const float* __restrict__ w2,  const float* __restrict__ b2,
             float* __restrict__ out, int n) {
  __shared__ __attribute__((aligned(16))) _Float16 ldsA[ROWS * LDS_STRIDE];
  __shared__ __attribute__((aligned(16))) _Float16 ldsH[ROWS * LDS_STRIDE];

  const int tid   = threadIdx.x;
  const int wave  = tid >> 5;
  const int lane  = tid & 31;
  const int m15   = lane & 15;
  const int khalf = lane >> 4;
  const int ncol  = wave * 16 + m15;   // this lane's output column in D tiles

  // Per-column BN fold: BN(v + b1) = v*s1 + o1 ; per-column bias for GEMM2.
  const float s1 = gamma[ncol] * rsqrtf(var[ncol] + BN_EPS);
  const float o1 = (b1[ncol] - mean[ncol]) * s1 + beta[ncol];
  const float o2 = b2[ncol];

  // Resident B fragments (K x N, lanes 0-15 -> K=0..15, lanes 16-31 -> K=16..31).
  v16h bw1[4], bw2[4];
#pragma unroll
  for (int kk = 0; kk < 4; ++kk) {
#pragma unroll
    for (int i = 0; i < 16; ++i) {
      int k = kk * 32 + khalf * 16 + i;
      bw1[kk][i] = (_Float16)w1[(size_t)k * DMID + ncol];
      bw2[kk][i] = (_Float16)w2[(size_t)k * DMID + ncol];
    }
  }

  const int ntiles = (n + ROWS - 1) / ROWS;
  const size_t tile_step = (size_t)gridDim.x * ROWS * DMID; // next-tile prefetch dist

  for (int tile = blockIdx.x; tile < ntiles; tile += gridDim.x) {
    const int rowbase = tile * ROWS;

    // ---- stage (in + agg) 32x128 tile -> LDS (f16); prefetch next tile ----
    {
      int seg = (tid & 15) * 8;    // 8-float segment within a row
#pragma unroll
      for (int rt = 0; rt < 2; ++rt) {
        int r    = rt * 16 + (tid >> 4);   // 0..31 row in tile
        int grow = rowbase + r;
        v8h hv = {};
        if (grow < n) {
          const float* ip = in  + (size_t)grow * DMID + seg;
          const float* gp = agg + (size_t)grow * DMID + seg;
          float4 a0 = *(const float4*)(ip);
          float4 a1 = *(const float4*)(ip + 4);
          float4 g0 = *(const float4*)(gp);
          float4 g1 = *(const float4*)(gp + 4);
          hv[0] = (_Float16)(a0.x + g0.x); hv[1] = (_Float16)(a0.y + g0.y);
          hv[2] = (_Float16)(a0.z + g0.z); hv[3] = (_Float16)(a0.w + g0.w);
          hv[4] = (_Float16)(a1.x + g1.x); hv[5] = (_Float16)(a1.y + g1.y);
          hv[6] = (_Float16)(a1.z + g1.z); hv[7] = (_Float16)(a1.w + g1.w);
          // Speculative prefetch of the next grid-stride tile (global_prefetch_b8)
          if ((size_t)grow * DMID + tile_step < (size_t)n * DMID) {
            __builtin_prefetch(ip + tile_step, 0, 0);
            __builtin_prefetch(gp + tile_step, 0, 0);
          }
        }
        *(v8h*)(&ldsA[r * LDS_STRIDE + seg]) = hv;
      }
    }
    __syncthreads();

    // ---- GEMM1 via WMMA (two independent 16x16 tiles), BN + ReLU -> ldsH ----
    v8f acc0 = {}, acc1 = {};
#pragma unroll
    for (int kk = 0; kk < 4; ++kk) {
      // A frag: lanes 0-15: K {kk*32+0..15}; lanes 16-31: {+16..31}
      const _Float16* r0 = &ldsA[m15 * LDS_STRIDE + kk * 32 + khalf * 16];
      const _Float16* r1 = r0 + 16 * LDS_STRIDE;
      v8h lo0 = *(const v8h*)(r0);
      v8h hi0 = *(const v8h*)(r0 + 8);
      v8h lo1 = *(const v8h*)(r1);
      v8h hi1 = *(const v8h*)(r1 + 8);
      v16h a0 = __builtin_shufflevector(lo0, hi0, 0,1,2,3,4,5,6,7,8,9,10,11,12,13,14,15);
      v16h a1 = __builtin_shufflevector(lo1, hi1, 0,1,2,3,4,5,6,7,8,9,10,11,12,13,14,15);
      acc0 = __builtin_amdgcn_wmma_f32_16x16x32_f16(false, a0, false, bw1[kk],
                                                    (short)0, acc0, false, false);
      acc1 = __builtin_amdgcn_wmma_f32_16x16x32_f16(false, a1, false, bw1[kk],
                                                    (short)0, acc1, false, false);
    }
#pragma unroll
    for (int r = 0; r < 8; ++r) {
      int mrow = r + khalf * 8;                 // D layout: M = r + 8*(lane>=16)
      float v0 = fmaxf(acc0[r] * s1 + o1, 0.0f);
      float v1 = fmaxf(acc1[r] * s1 + o1, 0.0f);
      ldsH[mrow * LDS_STRIDE + ncol]        = (_Float16)v0;
      ldsH[(mrow + 16) * LDS_STRIDE + ncol] = (_Float16)v1;
    }
    __syncthreads();

    // ---- GEMM2 via WMMA, + b2, ReLU, store ----
    v8f acc2 = {}, acc3 = {};
#pragma unroll
    for (int kk = 0; kk < 4; ++kk) {
      const _Float16* r0 = &ldsH[m15 * LDS_STRIDE + kk * 32 + khalf * 16];
      const _Float16* r1 = r0 + 16 * LDS_STRIDE;
      v8h lo0 = *(const v8h*)(r0);
      v8h hi0 = *(const v8h*)(r0 + 8);
      v8h lo1 = *(const v8h*)(r1);
      v8h hi1 = *(const v8h*)(r1 + 8);
      v16h a0 = __builtin_shufflevector(lo0, hi0, 0,1,2,3,4,5,6,7,8,9,10,11,12,13,14,15);
      v16h a1 = __builtin_shufflevector(lo1, hi1, 0,1,2,3,4,5,6,7,8,9,10,11,12,13,14,15);
      acc2 = __builtin_amdgcn_wmma_f32_16x16x32_f16(false, a0, false, bw2[kk],
                                                    (short)0, acc2, false, false);
      acc3 = __builtin_amdgcn_wmma_f32_16x16x32_f16(false, a1, false, bw2[kk],
                                                    (short)0, acc3, false, false);
    }
#pragma unroll
    for (int r = 0; r < 8; ++r) {
      int mrow = r + khalf * 8;
      int g0 = rowbase + mrow;
      int g1 = rowbase + 16 + mrow;
      if (g0 < n) out[(size_t)g0 * DMID + ncol] = fmaxf(acc2[r] + o2, 0.0f);
      if (g1 < n) out[(size_t)g1 * DMID + ncol] = fmaxf(acc3[r] + o2, 0.0f);
    }
    __syncthreads();
  }
}

// ---------------------------------------------------------------------------
// Batch pooling: g[batch[node]][coff + f] += h[node][f]
// ---------------------------------------------------------------------------
__global__ void gin_pool(const float* __restrict__ h, const int* __restrict__ batch,
                         float* __restrict__ g, int n, int coff) {
  size_t idx = (size_t)blockIdx.x * blockDim.x + threadIdx.x;
  if (idx >= (size_t)n * DMID) return;
  int node = (int)(idx >> 7);
  int f    = (int)(idx & (DMID - 1));
  int b    = batch[node];
  atomicAdd(&g[(size_t)b * (3 * DMID) + coff + f], h[idx]);
}

// ---------------------------------------------------------------------------
// Head: out = relu(g @ lin1 + b) @ lin2 + b2   (one block per graph, tiny)
// ---------------------------------------------------------------------------
__global__ void __launch_bounds__(384)
gin_head(const float* __restrict__ g,
         const float* __restrict__ l1w, const float* __restrict__ l1b,
         const float* __restrict__ l2w, const float* __restrict__ l2b,
         float* __restrict__ out) {
  __shared__ float sg[3 * DMID];
  __shared__ float sh[3 * DMID];
  const int b = blockIdx.x;
  const int j = threadIdx.x;
  sg[j] = g[(size_t)b * (3 * DMID) + j];
  __syncthreads();
  float acc = l1b[j];
  for (int k = 0; k < 3 * DMID; ++k)
    acc += sg[k] * l1w[(size_t)k * (3 * DMID) + j];
  sh[j] = fmaxf(acc, 0.0f);
  __syncthreads();
  if (j < 2) {
    float a = l2b[j];
    for (int k = 0; k < 3 * DMID; ++k)
      a += sh[k] * l2w[(size_t)k * 2 + j];
    out[b * 2 + j] = a;
  }
}

// ---------------------------------------------------------------------------
extern "C" void kernel_launch(void* const* d_in, const int* in_sizes, int n_in,
                              void* d_out, int out_size, void* d_ws, size_t ws_size,
                              hipStream_t stream) {
  const float* x     = (const float*)d_in[0];
  const int*   ei    = (const int*)d_in[1];
  const int*   batch = (const int*)d_in[2];
  const int n    = in_sizes[0] / DMID;
  const int numE = in_sizes[1] / 2;

  const float* P[3][8];   // w1, b1, gamma, beta, mean, var, w2, b2 per layer
  for (int l = 0; l < 3; ++l)
    for (int j = 0; j < 8; ++j)
      P[l][j] = (const float*)d_in[4 + l * 8 + j];
  const float* lin1_w = (const float*)d_in[28];
  const float* lin1_b = (const float*)d_in[29];
  const float* lin2_w = (const float*)d_in[30];
  const float* lin2_b = (const float*)d_in[31];

  // Workspace layout: agg | bufA | bufB | g
  size_t nf = (size_t)n * DMID;
  float* aggb = (float*)d_ws;
  float* bufA = aggb + nf;
  float* bufB = bufA + nf;
  float* g    = bufB + nf;

  hipMemsetAsync(g, 0, (size_t)NB * 3 * DMID * sizeof(float), stream);

  float* bufs[3] = { bufA, bufB, bufA };   // layer-3 reuses bufA
  const float* cur = x;
  const int sgrid = (numE + 7) / 8;                 // 8 edges per 256-thread block
  const int mgrid = 512;                            // grid-stride over 32-row tiles
  const int pgrid = (int)((nf + 255) / 256);

  for (int l = 0; l < 3; ++l) {
    hipMemsetAsync(aggb, 0, nf * sizeof(float), stream);
    gin_scatter_add<<<sgrid, 256, 0, stream>>>(cur, ei, aggb, numE);
    gin_mlp_wmma<<<mgrid, 256, 0, stream>>>(cur, aggb,
        P[l][0], P[l][1], P[l][2], P[l][3], P[l][4], P[l][5], P[l][6], P[l][7],
        bufs[l], n);
    gin_pool<<<pgrid, 256, 0, stream>>>(bufs[l], batch, g, n, l * DMID);
    cur = bufs[l];
  }
  gin_head<<<NB, 384, 0, stream>>>(g, lin1_w, lin1_b, lin2_w, lin2_b, (float*)d_out);
}
